// NVF_66331474919924
// MI455X (gfx1250) — compile-verified, matrix operand order checked by reading, not run on hardware
//
#include <hip/hip_runtime.h>

typedef __bf16 bf16_t;
typedef __attribute__((ext_vector_type(16))) __bf16 v16bf;
typedef __attribute__((ext_vector_type(8)))  __bf16 v8bf;
typedef __attribute__((ext_vector_type(8)))  float  v8f;

#define NB   4
#define NQ   4096
#define NN   4096
#define NK   8
#define HID  256
#define POSD 128
#define OUTD 128
#define CBN  128
#define NH   4
#define CDIM 64
#define MTOT (NB*NQ)             // 16384
#define F0IN ((OUTD+POSD)*NK)    // 2048
#define F0OUT (2*HID)            // 512

// ---------------- init scratch accumulators ----------------
__global__ void init_kernel(float* acc, int* flags)
{
  int t = threadIdx.x;
  if (t < 2)   acc[t] = 0.f;
  if (t < CBN) flags[t] = 0;
}

// ---------------- point encoder: pts = relu(x W1^T+b1) W2^T + b2 ----------------
__global__ __launch_bounds__(128)
void enc_kernel(const float* __restrict__ x,
                const float* __restrict__ w1, const float* __restrict__ b1,
                const float* __restrict__ w2, const float* __restrict__ b2,
                bf16_t* __restrict__ pts)
{
  __shared__ float h1[OUTD];
  int pt = blockIdx.x;                  // b*NN + n
  int j  = threadIdx.x;                 // output channel
  const float* xp = x + (size_t)pt * 3;
  float x0 = xp[0], x1 = xp[1], x2 = xp[2];
  float a = b1[j] + w1[j*3+0]*x0 + w1[j*3+1]*x1 + w1[j*3+2]*x2;
  h1[j] = fmaxf(a, 0.f);
  __syncthreads();
  float acc = b2[j];
  const float* wr = w2 + (size_t)j * OUTD;
  #pragma unroll 8
  for (int c = 0; c < OUTD; ++c) acc += wr[c] * h1[c];
  pts[(size_t)pt * OUTD + j] = (bf16_t)acc;   // no relu on layer 2
}

// ---------------- KNN: one wave per query, top-8 by squared distance ----------------
__global__ __launch_bounds__(256)
void knn_kernel(const float* __restrict__ q, const float* __restrict__ x,
                int* __restrict__ idx)
{
  int gw   = (blockIdx.x * blockDim.x + threadIdx.x) >> 5;   // query id (b*NQ+q)
  int lane = threadIdx.x & 31;
  int b = gw / NQ;
  const float* qp = q + (size_t)gw * 3;
  float qx = qp[0], qy = qp[1], qz = qp[2];
  const float* xb = x + (size_t)b * NN * 3;

  float d[8]; int id[8];
  #pragma unroll
  for (int s = 0; s < 8; ++s) { d[s] = 3.0e38f; id[s] = -1; }

  for (int n = lane; n < NN; n += 32) {
    const float* xp = xb + (size_t)n * 3;
    float dx = qx - xp[0], dy = qy - xp[1], dz = qz - xp[2];
    float dd = dx*dx + dy*dy + dz*dz;
    int ii = n;
    #pragma unroll
    for (int s = 0; s < 8; ++s) {       // bubble-insert, static indexing only
      if (dd < d[s]) { float td=d[s]; int ti=id[s]; d[s]=dd; id[s]=ii; dd=td; ii=ti; }
    }
  }
  // merge across lanes: 8 rounds of wave-min + pop winner
  #pragma unroll
  for (int r = 0; r < 8; ++r) {
    float bd = d[0]; int bidx = id[0];
    #pragma unroll
    for (int off = 16; off; off >>= 1) {
      float od = __shfl_xor(bd, off, 32);
      int   oi = __shfl_xor(bidx, off, 32);
      if (od < bd || (od == bd && (unsigned)oi < (unsigned)bidx)) { bd = od; bidx = oi; }
    }
    bool won = (id[0] == bidx) && (d[0] == bd);
    if (won) {
      #pragma unroll
      for (int s = 0; s < 7; ++s) { d[s] = d[s+1]; id[s] = id[s+1]; }
      d[7] = 3.0e38f; id[7] = -1;
    }
    if (lane == 0) idx[(size_t)gw * NK + r] = bidx;
  }
}

// ---------------- gather + fc_pos + feats assembly (bf16) ----------------
__global__ __launch_bounds__(128)
void feats_kernel(const float* __restrict__ q, const float* __restrict__ x,
                  const int* __restrict__ idx, const bf16_t* __restrict__ pts,
                  const float* __restrict__ pw, const float* __restrict__ pb,
                  bf16_t* __restrict__ feats)
{
  int w = blockIdx.x;            // m*NK + k
  int m = w >> 3;
  int k = w & 7;
  int j = threadIdx.x;           // 0..127
  int b = m / NQ;
  int nid = idx[(size_t)m * NK + k];
  const float* qp = q + (size_t)m * 3;
  const float* xp = x + ((size_t)b * NN + nid) * 3;
  float pi[9];
  pi[0]=qp[0]; pi[1]=qp[1]; pi[2]=qp[2];
  pi[3]=xp[0]; pi[4]=xp[1]; pi[5]=xp[2];
  pi[6]=pi[3]-pi[0]; pi[7]=pi[4]-pi[1]; pi[8]=pi[5]-pi[2];
  const float* wr = pw + (size_t)j * 9;
  float acc = pb[j];
  #pragma unroll
  for (int c = 0; c < 9; ++c) acc += wr[c] * pi[c];
  bf16_t* fr = feats + (size_t)m * F0IN + (size_t)k * (POSD + OUTD);
  fr[j]        = (bf16_t)fmaxf(acc, 0.f);                       // posf
  fr[POSD + j] = pts[((size_t)b * NN + nid) * OUTD + j];        // qnkf
}

// ---------------- f32 -> bf16 weight convert ----------------
__global__ void cvt_kernel(const float* __restrict__ s, bf16_t* __restrict__ dst, int n)
{
  int i = blockIdx.x * blockDim.x + threadIdx.x;
  if (i < n) dst[i] = (bf16_t)s[i];
}

// ---------------- WMMA GEMM: C[m,n] = relu(sum_k A[m,k] W[n,k] + bias[n]) ----------------
// one wave per 16x16 tile, K step 32, v_wmma_f32_16x16x32_bf16
__global__ __launch_bounds__(256)
void gemm_kernel(const bf16_t* __restrict__ A, int lda,
                 const bf16_t* __restrict__ W,
                 const float* __restrict__ bias,
                 bf16_t* __restrict__ C, int ldc,
                 int M, int N, int K, int do_relu)
{
  int wid  = (blockIdx.x * blockDim.x + threadIdx.x) >> 5;
  int lane = threadIdx.x & 31;
  int ntn = N >> 4;
  if (wid >= (M >> 4) * ntn) return;
  int tm = (wid / ntn) << 4;
  int tn = (wid % ntn) << 4;
  int hi = lane >> 4;         // half-wave select (K sub-block)
  int lm = lane & 15;         // row (A) / col (B)
  const bf16_t* arow = A + (size_t)(tm + lm) * lda;
  const bf16_t* wrow = W + (size_t)(tn + lm) * K;
  v8f acc = {};
  for (int kt = 0; kt < K; kt += 32) {
    __builtin_prefetch(arow + kt + 64, 0, 1);
    __builtin_prefetch(wrow + kt + 64, 0, 1);
    int ka = kt + hi * 8;
    v8bf a0 = *reinterpret_cast<const v8bf*>(arow + ka);
    v8bf a1 = *reinterpret_cast<const v8bf*>(arow + ka + 16);
    v8bf b0 = *reinterpret_cast<const v8bf*>(wrow + ka);
    v8bf b1 = *reinterpret_cast<const v8bf*>(wrow + ka + 16);
    v16bf af, bfm;
    #pragma unroll
    for (int e = 0; e < 8; ++e) {
      af[e]  = a0[e]; af[e+8]  = a1[e];
      bfm[e] = b0[e]; bfm[e+8] = b1[e];
    }
    acc = __builtin_amdgcn_wmma_f32_16x16x32_bf16(false, af, false, bfm,
                                                  (short)0, acc, false, false);
  }
  int n = tn + lm;
  float bv = bias[n];
  #pragma unroll
  for (int r = 0; r < 8; ++r) {
    int mrow = tm + r + hi * 8;
    float v = acc[r] + bv;
    if (do_relu) v = fmaxf(v, 0.f);
    C[(size_t)mrow * ldc + n] = (bf16_t)v;
  }
}

// ---------------- VQ: one wave per (m,h); argmin over 128 codes, write quant ----------------
__global__ __launch_bounds__(256)
void vq_kernel(bf16_t* __restrict__ fbuf /* [MTOT,512]: [f | fvq] */,
               const float* __restrict__ cb,
               float* __restrict__ acc, int* __restrict__ flags)
{
  int gw   = (blockIdx.x * blockDim.x + threadIdx.x) >> 5;  // 0..MTOT*NH-1
  int lane = threadIdx.x & 31;
  int wl   = (threadIdx.x >> 5) & 7;
  int m = gw >> 2;
  int h = gw & 3;
  __shared__ float sxh[8][CDIM];
  const bf16_t* xrow = fbuf + (size_t)m * (2*HID) + h * CDIM;
  sxh[wl][lane]      = (float)xrow[lane];
  sxh[wl][lane + 32] = (float)xrow[lane + 32];
  __syncthreads();
  const float* cbh = cb + (size_t)h * CBN * CDIM;
  float best = 3.0e38f; int bi = 0;
  for (int c = lane; c < CBN; c += 32) {
    const float* cr = cbh + (size_t)c * CDIM;
    float s = 0.f;
    #pragma unroll 8
    for (int d = 0; d < CDIM; ++d) { float df = sxh[wl][d] - cr[d]; s += df*df; }
    if (s < best) { best = s; bi = c; }
  }
  #pragma unroll
  for (int off = 16; off; off >>= 1) {
    float ob = __shfl_xor(best, off, 32);
    int   oi = __shfl_xor(bi,   off, 32);
    if (ob < best || (ob == best && oi < bi)) { best = ob; bi = oi; }
  }
  if (lane == 0) {
    atomicAdd(acc, best);                 // commit-loss partial: min dist == ||xh-quant||^2
    if (h == 0) flags[bi] = 1;            // usage over ind[0]
  }
  bf16_t* qrow = fbuf + (size_t)m * (2*HID) + HID + h * CDIM;   // fvq half
  const float* cw = cbh + (size_t)bi * CDIM;
  qrow[lane]      = (bf16_t)cw[lane];
  qrow[lane + 32] = (bf16_t)cw[lane + 32];
}

// ---------------- codebook squared norms (one wave per (h,i)) ----------------
__global__ __launch_bounds__(256)
void normsq_kernel(const float* __restrict__ cb, float* __restrict__ nsq)
{
  int gw   = (blockIdx.x * blockDim.x + threadIdx.x) >> 5;   // 0..NH*CBN-1
  int lane = threadIdx.x & 31;
  const float* cr = cb + (size_t)gw * CDIM;
  float a0 = cr[lane], a1 = cr[lane + 32];
  float s = a0*a0 + a1*a1;
  #pragma unroll
  for (int off = 16; off; off >>= 1) s += __shfl_xor(s, off, 32);
  if (lane == 0) nsq[gw] = s;
}

// ---------------- ortho loss: sum_{h,i,j} cos^2(c_i,c_j) ----------------
__global__ __launch_bounds__(128)
void ortho_kernel(const float* __restrict__ cb, const float* __restrict__ nsq,
                  float* __restrict__ acc)
{
  int h = blockIdx.x / CBN;
  int i = blockIdx.x % CBN;
  int j = threadIdx.x;                 // 0..127
  __shared__ float ci[CDIM];
  __shared__ float red[128];
  const float* cir = cb + ((size_t)h * CBN + i) * CDIM;
  if (j < CDIM) ci[j] = cir[j];
  __syncthreads();
  const float* cjr = cb + ((size_t)h * CBN + j) * CDIM;
  float dot = 0.f;
  #pragma unroll 8
  for (int d = 0; d < CDIM; ++d) dot += ci[d] * cjr[d];
  float val = dot * dot / (nsq[h * CBN + i] * nsq[h * CBN + j]);
  red[j] = val;
  __syncthreads();
  for (int s = 64; s > 0; s >>= 1) {
    if (j < s) red[j] += red[j + s];
    __syncthreads();
  }
  if (j == 0) atomicAdd(acc + 1, red[0]);
}

// ---------------- output head: 256 -> 3, f32 ----------------
__global__ __launch_bounds__(256)
void fcout_kernel(const bf16_t* __restrict__ net, const float* __restrict__ w,
                  const float* __restrict__ b, float* __restrict__ out)
{
  int m = blockIdx.x * blockDim.x + threadIdx.x;
  if (m >= MTOT) return;
  const bf16_t* nr = net + (size_t)m * HID;
  float a0 = b[0], a1 = b[1], a2 = b[2];
  for (int c = 0; c < HID; ++c) {
    float v = (float)nr[c];
    a0 += w[0*HID + c] * v;
    a1 += w[1*HID + c] * v;
    a2 += w[2*HID + c] * v;
  }
  out[(size_t)m*3+0] = a0; out[(size_t)m*3+1] = a1; out[(size_t)m*3+2] = a2;
}

// ---------------- scalars: loss_vq, usage ----------------
__global__ void finalize_kernel(const float* __restrict__ acc,
                                const int* __restrict__ flags,
                                float* __restrict__ tail)
{
  if (threadIdx.x == 0) {
    float commit = 0.001f * acc[0] / (float)((size_t)NH * NB * NQ * CDIM);
    float ortho  = 1e-5f * (acc[1] / (float)(NH * CBN * CBN) - 1.0f / (float)CBN);
    tail[0] = commit + ortho;
    int cnt = 0;
    for (int i = 0; i < CBN; ++i) cnt += flags[i];
    tail[1] = (float)cnt / (float)CBN;
  }
}

extern "C" void kernel_launch(void* const* d_in, const int* in_sizes, int n_in,
                              void* d_out, int out_size, void* d_ws, size_t ws_size,
                              hipStream_t stream)
{
  const float* q        = (const float*)d_in[0];
  const float* x        = (const float*)d_in[1];
  const float* enc_w1   = (const float*)d_in[2];
  const float* enc_b1   = (const float*)d_in[3];
  const float* enc_w2   = (const float*)d_in[4];
  const float* enc_b2   = (const float*)d_in[5];
  const float* fc_pos_w = (const float*)d_in[6];
  const float* fc_pos_b = (const float*)d_in[7];
  const float* fc0_w = (const float*)d_in[8];
  const float* fc0_b = (const float*)d_in[9];
  const float* fc1_w = (const float*)d_in[10];
  const float* fc1_b = (const float*)d_in[11];
  const float* fc2_w = (const float*)d_in[12];
  const float* fc2_b = (const float*)d_in[13];
  const float* fc3_w = (const float*)d_in[14];
  const float* fc3_b = (const float*)d_in[15];
  const float* fc4_w = (const float*)d_in[16];
  const float* fc4_b = (const float*)d_in[17];
  const float* fcout_w = (const float*)d_in[18];
  const float* fcout_b = (const float*)d_in[19];
  const float* codebook = (const float*)d_in[20];
  float* out = (float*)d_out;

  char* ws = (char*)d_ws;
  size_t off = 0;
  auto take = [&](size_t bytes) -> char* {
    char* p = ws + off;
    off += (bytes + 255) & ~(size_t)255;
    return p;
  };
  bf16_t* pts   = (bf16_t*)take((size_t)NB*NN*OUTD*2);     // 4 MiB
  int*    idx   = (int*)   take((size_t)MTOT*NK*4);        // 512 KiB
  bf16_t* feats = (bf16_t*)take((size_t)MTOT*F0IN*2);      // 64 MiB
  bf16_t* w0 = (bf16_t*)take((size_t)F0OUT*F0IN*2);
  bf16_t* w1 = (bf16_t*)take((size_t)HID*F0OUT*2);
  bf16_t* w2 = (bf16_t*)take((size_t)HID*HID*2);
  bf16_t* w3 = (bf16_t*)take((size_t)HID*F0OUT*2);
  bf16_t* w4 = (bf16_t*)take((size_t)HID*HID*2);
  bf16_t* f0   = (bf16_t*)take((size_t)MTOT*F0OUT*2);      // 16 MiB
  bf16_t* f1   = (bf16_t*)take((size_t)MTOT*HID*2);        // 8 MiB
  bf16_t* f3in = (bf16_t*)take((size_t)MTOT*2*HID*2);      // 16 MiB: [f | fvq]
  bf16_t* f3   = (bf16_t*)take((size_t)MTOT*HID*2);
  bf16_t* net  = (bf16_t*)take((size_t)MTOT*HID*2);
  float*  nsq   = (float*)take((size_t)NH*CBN*4);
  float*  acc   = (float*)take(2*4);
  int*    flags = (int*)take(CBN*4);

  init_kernel<<<1, 256, 0, stream>>>(acc, flags);
  enc_kernel<<<NB*NN, 128, 0, stream>>>(x, enc_w1, enc_b1, enc_w2, enc_b2, pts);
  knn_kernel<<<(MTOT*32)/256, 256, 0, stream>>>(q, x, idx);
  feats_kernel<<<MTOT*NK, 128, 0, stream>>>(q, x, idx, pts, fc_pos_w, fc_pos_b, feats);
  cvt_kernel<<<(F0OUT*F0IN+255)/256, 256, 0, stream>>>(fc0_w, w0, F0OUT*F0IN);
  cvt_kernel<<<(HID*F0OUT+255)/256, 256, 0, stream>>>(fc1_w, w1, HID*F0OUT);
  cvt_kernel<<<(HID*HID+255)/256, 256, 0, stream>>>(fc2_w, w2, HID*HID);
  cvt_kernel<<<(HID*F0OUT+255)/256, 256, 0, stream>>>(fc3_w, w3, HID*F0OUT);
  cvt_kernel<<<(HID*HID+255)/256, 256, 0, stream>>>(fc4_w, w4, HID*HID);

  auto gemm = [&](const bf16_t* A, int lda, const bf16_t* W, const float* bias,
                  bf16_t* C, int ldc, int M, int N, int K, int relu) {
    int waves = (M >> 4) * (N >> 4);
    gemm_kernel<<<(waves + 7) / 8, 256, 0, stream>>>(A, lda, W, bias, C, ldc, M, N, K, relu);
  };
  gemm(feats, F0IN,  w0, fc0_b, f0,   F0OUT, MTOT, F0OUT, F0IN,  1);  // fc0
  gemm(f0,    F0OUT, w1, fc1_b, f1,   HID,   MTOT, HID,   F0OUT, 1);  // fc1
  gemm(f1,    HID,   w2, fc2_b, f3in, 2*HID, MTOT, HID,   HID,   1);  // fc2 -> left half of f3in

  vq_kernel<<<(MTOT*NH*32)/256, 256, 0, stream>>>(f3in, codebook, acc, flags);
  normsq_kernel<<<(NH*CBN*32)/256, 256, 0, stream>>>(codebook, nsq);
  ortho_kernel<<<NH*CBN, 128, 0, stream>>>(codebook, nsq, acc);

  gemm(f3in, 2*HID, w3, fc3_b, f3,  HID, MTOT, HID, 2*HID, 1);        // fc3
  gemm(f3,   HID,   w4, fc4_b, net, HID, MTOT, HID, HID,   1);        // fc4

  fcout_kernel<<<(MTOT+255)/256, 256, 0, stream>>>(net, fcout_w, fcout_b, out);
  finalize_kernel<<<1, 64, 0, stream>>>(acc, flags, out + (size_t)MTOT*3);

  (void)in_sizes; (void)n_in; (void)out_size; (void)ws_size;
}
